// Seq2Seq_32255204393212
// MI455X (gfx1250) — matrix operand order repaired and checked by reading, not compiled, not run on hardware
//
#include <hip/hip_runtime.h>

typedef __attribute__((ext_vector_type(16))) _Float16 v16h;
typedef __attribute__((ext_vector_type(8)))  _Float16 v8h;
typedef __attribute__((ext_vector_type(8)))  float    v8f;

static constexpr int B_    = 256;
static constexpr int H_    = 256;
static constexpr int G_    = 1024;   // 4*H
static constexpr int E_    = 256;
static constexpr int TIN_  = 32;
static constexpr int TOUT_ = 29;
static constexpr int D_    = 6;
static constexpr int VDEC_ = 72;

// ---------------------------------------------------------------------------
// WMMA fragment loaders (CDNA5 wave32 layouts, CDNA5 ISA §7.12.2)
// A: 16x32 f16, lane L holds row M=L&15; lanes 0-15 K∈{0..7,16..23},
//    lanes 16-31 K∈{8..15,24..31}  -> two contiguous 16B chunks per lane.
// B: 32x16 f16, lane L holds col N=L&15; lanes 0-15 K∈{0..7,16..23},
//    lanes 16-31 K∈{8..15,24..31} -> two contiguous 16B chunks per lane (row of W[N,K]).
// ---------------------------------------------------------------------------
__device__ __forceinline__ v16h load_a_frag(const _Float16* __restrict__ base,
                                            int ld, int m0, int k0, int lane) {
  const int m  = m0 + (lane & 15);
  const int bk = (lane & 16) ? 8 : 0;
  const _Float16* p = base + (size_t)m * ld + (k0 + bk);
  union { v16h v; v8h h[2]; } u;
  u.h[0] = *(const v8h*)(p);
  u.h[1] = *(const v8h*)(p + 16);
  return u.v;
}

__device__ __forceinline__ v16h load_b_frag(const _Float16* __restrict__ w,
                                            int ld, int n0, int k0, int lane) {
  const int n  = n0 + (lane & 15);
  const int bk = (lane & 16) ? 8 : 0;
  const _Float16* p = w + (size_t)n * ld + (k0 + bk);
  union { v16h v; v8h h[2]; } u;
  u.h[0] = *(const v8h*)(p);
  u.h[1] = *(const v8h*)(p + 16);
  return u.v;
}

// 32x64 output strip per wave: 2 A-frags reuse 4 B-frags -> 8 WMMA / K-step.
__device__ __forceinline__ void gemm_core32(const _Float16* __restrict__ X, int K,
                                            const _Float16* __restrict__ W,
                                            int m0, int n0, int lane,
                                            v8f (&acc)[2][4]) {
  for (int k0 = 0; k0 < K; k0 += 32) {
    v16h a[2];
#pragma unroll
    for (int i = 0; i < 2; ++i) a[i] = load_a_frag(X, K, m0 + 16 * i, k0, lane);
#pragma unroll
    for (int j = 0; j < 4; ++j) {
      const v16h b = load_b_frag(W, K, n0 + 16 * j, k0, lane);
#pragma unroll
      for (int i = 0; i < 2; ++i)
        acc[i][j] = __builtin_amdgcn_wmma_f32_16x16x32_f16(
            false, a[i], false, b, (short)0, acc[i][j], false, false);
    }
  }
}

__device__ __forceinline__ float sigf(float x) {
  return 1.0f / (1.0f + __expf(-x));
}

// ---------------------------------------------------------------------------
// Fused bidirectional-LSTM layer kernel:
//   gates = X*WxP^T + H16*WhP^T  (weights row-PERMUTED so each 64-col strip is
//   [i|f|g|o] tiles of the same 16 hidden units), then the LSTM nonlinearity
//   is applied in-register and c/h/ybuf written directly. grid.z = direction.
//   h16 input/output are DIFFERENT buffers (double-buffered by caller).
// ---------------------------------------------------------------------------
__global__ __launch_bounds__(128)
void wmma_lstm_layer(const _Float16* __restrict__ X, int Kx,
                     const _Float16* __restrict__ WxP, long wxZ,
                     const _Float16* __restrict__ H16, long hZ,
                     const _Float16* __restrict__ WhP, long whZ,
                     const float* __restrict__ bias, long biasZ,  // orig i,f,g,o layout
                     float* __restrict__ c32,                      // slice [2,B,H]
                     float* __restrict__ h32,                      // nullable
                     _Float16* __restrict__ h16o,
                     _Float16* __restrict__ c16o,                  // nullable
                     _Float16* __restrict__ ybuf,                  // [B,2H]
                     long stZ) {
  const int z = blockIdx.z;
  const _Float16* Wx = WxP + (size_t)z * wxZ;
  const _Float16* Hs = H16 + (size_t)z * hZ;
  const _Float16* Wh = WhP + (size_t)z * whZ;
  const float*    bb = bias + (size_t)z * biasZ;
  const size_t    so = (size_t)z * stZ;

  const int wave = (int)((blockIdx.x * blockDim.x + threadIdx.x) >> 5);
  const int lane = (int)(threadIdx.x & 31);
  const int mt = wave >> 4;          // 8 m-tiles of 32 batch rows
  const int nt = wave & 15;          // 16 strips of 16 hidden units
  const int m0 = mt * 32, n0 = nt * 64;

  v8f acc[2][4] = {};
  gemm_core32(X,  Kx, Wx, m0, n0, lane, acc);
  gemm_core32(Hs, H_, Wh, m0, n0, lane, acc);

  // lane owns hidden unit j for 2x8 batch rows; acc[i][q] = gate q (i,f,g,o)
  const int j = nt * 16 + (lane & 15);
  const float bi = bb[0 * H_ + j], bf = bb[1 * H_ + j];
  const float bg = bb[2 * H_ + j], bo = bb[3 * H_ + j];
#pragma unroll
  for (int i = 0; i < 2; ++i) {
    const int mrb = m0 + 16 * i + ((lane & 16) ? 8 : 0);
#pragma unroll
    for (int v = 0; v < 8; ++v) {
      const int b = mrb + v;
      const size_t idx = so + (size_t)b * H_ + j;
      const float ig = sigf(acc[i][0][v] + bi);
      const float fg = sigf(acc[i][1][v] + bf);
      const float gt = tanhf(acc[i][2][v] + bg);
      const float og = sigf(acc[i][3][v] + bo);
      const float cn = fg * c32[idx] + ig * gt;
      const float hn = og * tanhf(cn);
      c32[idx]  = cn;
      h16o[idx] = (_Float16)hn;
      if (h32)  h32[idx]  = hn;
      if (c16o) c16o[idx] = (_Float16)cn;
      ybuf[(size_t)b * (2 * H_) + (size_t)z * H_ + j] = (_Float16)hn;
    }
  }
}

// ---------------------------------------------------------------------------
// Generic WMMA GEMM: out = X[M,K] * W[Npad,K]^T (+bias), 32x64 per wave.
// ---------------------------------------------------------------------------
__global__ __launch_bounds__(128)
void wmma_gemm(const _Float16* __restrict__ X,
               const _Float16* __restrict__ W, int K,
               float* __restrict__ out, int ldo,
               _Float16* __restrict__ out16,
               const float* __restrict__ bias,
               int M, int Nreal, int Nt64) {
  const int wave = (int)((blockIdx.x * blockDim.x + threadIdx.x) >> 5);
  const int lane = (int)(threadIdx.x & 31);
  const int mt = wave / Nt64;
  const int nt = wave - mt * Nt64;
  if (mt * 32 >= M) return;            // wave-uniform
  const int m0 = mt * 32, n0 = nt * 64;

  v8f acc[2][4] = {};
  gemm_core32(X, K, W, m0, n0, lane, acc);

#pragma unroll
  for (int jt = 0; jt < 4; ++jt) {
    const int n = n0 + jt * 16 + (lane & 15);
    if (n < Nreal) {
      const float bv = bias ? bias[n] : 0.0f;
#pragma unroll
      for (int i = 0; i < 2; ++i) {
        const int mrb = m0 + 16 * i + ((lane & 16) ? 8 : 0);
#pragma unroll
        for (int v = 0; v < 8; ++v) {
          const float val = acc[i][jt][v] + bv;
          const size_t off = (size_t)(mrb + v) * ldo + n;
          if (out)   out[off]   = val;
          if (out16) out16[off] = (_Float16)val;
        }
      }
    }
  }
}

// out[m,b,h] = sum_l W2[m,l]*hn[l,b,h] + b2[m]   (mix over the D=6 axis)
__global__ void mix_dirs(const float* __restrict__ hn, const float* __restrict__ W2,
                         const float* __restrict__ b2, float* __restrict__ out,
                         _Float16* __restrict__ out16) {
  const int t = blockIdx.x * blockDim.x + threadIdx.x;
  if (t >= B_ * H_) return;
  const int b = t / H_;
  const int j = t - b * H_;
  float v[D_];
#pragma unroll
  for (int l = 0; l < D_; ++l) v[l] = hn[((size_t)l * B_ + b) * H_ + j];
#pragma unroll
  for (int m = 0; m < D_; ++m) {
    float s = b2[m];
#pragma unroll
    for (int l = 0; l < D_; ++l) s += W2[m * D_ + l] * v[l];
    const size_t idx = ((size_t)m * B_ + b) * H_ + j;
    out[idx] = s;
    if (out16) out16[idx] = (_Float16)s;
  }
}

// Per-(b,h) softmax over Tin per direction, weighted sum, mean over D.
__global__ void attn_ctx(const float* __restrict__ dech,
                         const float* __restrict__ encout,
                         float* __restrict__ ctx) {
  const int t = blockIdx.x * blockDim.x + threadIdx.x;
  if (t >= B_ * H_) return;
  const int b = t / H_;
  const int j = t - b * H_;
  float acc = 0.0f;
#pragma unroll
  for (int d = 0; d < D_; ++d) {
    const float hv = dech[((size_t)d * B_ + b) * H_ + j];
    float ev[TIN_];
    float mx = -1e30f;
#pragma unroll
    for (int ti = 0; ti < TIN_; ++ti) {
      ev[ti] = encout[(((size_t)ti * D_ + d) * B_ + b) * H_ + j];
      mx = fmaxf(mx, hv * ev[ti]);
    }
    float s = 0.0f, a = 0.0f;
#pragma unroll
    for (int ti = 0; ti < TIN_; ++ti) {
      const float w = __expf(hv * ev[ti] - mx);
      s += w;
      a += w * ev[ti];
    }
    acc += a / s;
  }
  ctx[(size_t)b * H_ + j] = acc * (1.0f / (float)D_);
}

__global__ void embed_tokens(const int* __restrict__ tok, int tokStride,
                             const float* __restrict__ emb,
                             _Float16* __restrict__ x) {
  const int t = blockIdx.x * blockDim.x + threadIdx.x;
  if (t >= B_ * E_) return;
  const int b = t / E_;
  const int e = t - b * E_;
  const int id = tok[(size_t)b * tokStride];
  x[(size_t)b * E_ + e] = (_Float16)emb[(size_t)id * E_ + e];
}

__global__ void build_dec_input(const int* __restrict__ target, int step,
                                const float* __restrict__ demb,
                                const float* __restrict__ ctx,
                                _Float16* __restrict__ x) {
  const int t = blockIdx.x * blockDim.x + threadIdx.x;
  if (t >= B_ * 512) return;
  const int b = t / 512;
  const int k = t & 511;
  float v;
  if (k < E_) {
    const int id = target[(size_t)b * TOUT_ + step];
    v = demb[(size_t)id * E_ + k];
  } else {
    v = ctx[(size_t)b * H_ + (k - E_)];
  }
  x[t] = (_Float16)v;
}

// Plain f32->f16 with optional zero row padding.
__global__ void cvt_pad(const float* __restrict__ src, _Float16* __restrict__ dst,
                        int srcRows, int cols, long total) {
  const long t = (long)blockIdx.x * blockDim.x + threadIdx.x;
  if (t >= total) return;
  const long r = t / cols;
  dst[t] = (r < srcRows) ? (_Float16)src[t] : (_Float16)0.0f;
}

// Gate-weight conversion with row permutation: within each G-row block,
// dst row n = 64*s + 16*q + jj  <-  src row q*H + 16*s + jj, so that a 64-wide
// WMMA strip holds [i|f|g|o] tiles of the same 16 hidden units.
__global__ void cvt_perm_gates(const float* __restrict__ src, _Float16* __restrict__ dst,
                               int cols, long total) {
  const long t = (long)blockIdx.x * blockDim.x + threadIdx.x;
  if (t >= total) return;
  const long r = t / cols;
  const int  c = (int)(t - r * cols);
  const long blk = r / G_;
  const int  rr  = (int)(r - blk * G_);
  const int  s = rr >> 6, q = (rr >> 4) & 3, jj = rr & 15;
  const long srcr = blk * G_ + q * H_ + s * 16 + jj;
  dst[t] = (_Float16)src[srcr * cols + c];
}

__global__ void zero_f32(float* p, long n) {
  const long t = (long)blockIdx.x * blockDim.x + threadIdx.x;
  if (t < n) p[t] = 0.0f;
}
__global__ void zero_f16(_Float16* p, long n) {
  const long t = (long)blockIdx.x * blockDim.x + threadIdx.x;
  if (t < n) p[t] = (_Float16)0.0f;
}
__global__ void set_pred0(float* out) {
  const int b = blockIdx.x * blockDim.x + threadIdx.x;
  if (b < B_) out[(size_t)b * VDEC_ + 1] = 1.0f;
}

// ---------------------------------------------------------------------------
extern "C" void kernel_launch(void* const* d_in, const int* in_sizes, int n_in,
                              void* d_out, int out_size, void* d_ws, size_t ws_size,
                              hipStream_t stream) {
  (void)in_sizes; (void)n_in; (void)ws_size;
  const int*   input    = (const int*)  d_in[0];
  const int*   target   = (const int*)  d_in[1];
  const float* enc_emb  = (const float*)d_in[3];
  const float* dec_emb  = (const float*)d_in[4];
  const float* enc_Wih0 = (const float*)d_in[5];
  const float* enc_Wih1 = (const float*)d_in[6];
  const float* enc_Whh  = (const float*)d_in[7];
  const float* enc_b    = (const float*)d_in[8];
  const float* dec_Wih0 = (const float*)d_in[9];
  const float* dec_Wih1 = (const float*)d_in[10];
  const float* dec_Whh  = (const float*)d_in[11];
  const float* dec_b    = (const float*)d_in[12];
  const float* lh_W1 = (const float*)d_in[13]; const float* lh_b1 = (const float*)d_in[14];
  const float* lh_W2 = (const float*)d_in[15]; const float* lh_b2 = (const float*)d_in[16];
  const float* lc_W1 = (const float*)d_in[17]; const float* lc_b1 = (const float*)d_in[18];
  const float* lc_W2 = (const float*)d_in[19]; const float* lc_b2 = (const float*)d_in[20];
  const float* la_W1 = (const float*)d_in[21]; const float* la_b1 = (const float*)d_in[22];
  const float* la_W2 = (const float*)d_in[23]; const float* la_b2 = (const float*)d_in[24];
  const float* attn_W = (const float*)d_in[25]; const float* attn_b = (const float*)d_in[26];
  const float* fc_W   = (const float*)d_in[27]; const float* fc_b   = (const float*)d_in[28];
  float* out = (float*)d_out;

  // ---- workspace bump allocator ----
  char* wp = (char*)d_ws;
  auto alloc = [&](size_t bytes) {
    char* r = wp;
    wp += (bytes + 255) & ~(size_t)255;
    return r;
  };
  auto a16 = [&](size_t n) { return (_Float16*)alloc(n * sizeof(_Float16)); };
  auto a32 = [&](size_t n) { return (float*)alloc(n * sizeof(float)); };

  _Float16* eWih0_16 = a16((size_t)2 * G_ * E_);     // gate-permuted
  _Float16* eWih1_16 = a16((size_t)4 * G_ * 512);
  _Float16* eWhh_16  = a16((size_t)6 * G_ * H_);
  _Float16* dWih0_16 = a16((size_t)2 * G_ * E_);
  _Float16* dWih1_16 = a16((size_t)4 * G_ * 512);
  _Float16* dWhh_16  = a16((size_t)6 * G_ * H_);
  _Float16* attnW16  = a16((size_t)E_ * 512);
  _Float16* fcW16    = a16((size_t)128 * 512);       // padded 72 -> 128 rows
  _Float16* lhW1_16  = a16((size_t)H_ * H_);
  _Float16* lcW1_16  = a16((size_t)H_ * H_);
  _Float16* laW1_16  = a16((size_t)H_ * H_);

  float* encout = a32((size_t)TIN_ * D_ * B_ * H_);  // [Tin,D,B,H]  50 MB
  float* cenc   = a32((size_t)D_ * B_ * H_);
  float* hdec   = a32((size_t)D_ * B_ * H_);
  float* cdec   = a32((size_t)D_ * B_ * H_);
  float* hn     = a32((size_t)D_ * B_ * H_);
  float* ctx    = a32((size_t)B_ * H_);
  _Float16* henc16[2] = { a16((size_t)D_ * B_ * H_), a16((size_t)D_ * B_ * H_) };
  _Float16* hdec16[2] = { a16((size_t)D_ * B_ * H_), a16((size_t)D_ * B_ * H_) };
  _Float16* cenc16 = a16((size_t)D_ * B_ * H_);
  _Float16* x16    = a16((size_t)B_ * E_);
  _Float16* ybuf   = a16((size_t)B_ * 512);
  _Float16* xatt16 = a16((size_t)B_ * E_);
  _Float16* xcat16 = a16((size_t)B_ * 512);

  const long BH = (long)B_ * H_;

  auto gemm = [&](const _Float16* X, const _Float16* W, int K,
                  float* o, int ldo, _Float16* o16, const float* bias,
                  int M, int Nreal, int Npad) {
    const int Nt64  = Npad / 64;
    const int waves = (M / 32) * Nt64;
    wmma_gemm<<<dim3((unsigned)((waves + 3) / 4)), 128, 0, stream>>>(
        X, W, K, o, ldo, o16, bias, M, Nreal, Nt64);
  };
  auto cvt = [&](const float* s, _Float16* d, int srcRows, int cols, int dstRows) {
    const long total = (long)dstRows * cols;
    cvt_pad<<<(unsigned)((total + 255) / 256), 256, 0, stream>>>(s, d, srcRows, cols, total);
  };
  auto cvtg = [&](const float* s, _Float16* d, int blocks, int cols) {
    const long total = (long)blocks * G_ * cols;
    cvt_perm_gates<<<(unsigned)((total + 255) / 256), 256, 0, stream>>>(s, d, cols, total);
  };

  // ---- weight conversion (f32 -> f16; gate weights row-permuted) ----
  cvtg(enc_Wih0, eWih0_16, 2, E_);
  cvtg(enc_Wih1, eWih1_16, 4, 512);
  cvtg(enc_Whh,  eWhh_16,  6, H_);
  cvtg(dec_Wih0, dWih0_16, 2, E_);
  cvtg(dec_Wih1, dWih1_16, 4, 512);
  cvtg(dec_Whh,  dWhh_16,  6, H_);
  cvt(attn_W, attnW16, E_, 512, E_);
  cvt(fc_W,   fcW16,   VDEC_, 512, 128);
  cvt(lh_W1, lhW1_16, H_, H_, H_);
  cvt(lc_W1, lcW1_16, H_, H_, H_);
  cvt(la_W1, laW1_16, H_, H_, H_);

  // ---- init states & output ----
  const long st = (long)D_ * B_ * H_;
  zero_f32<<<(unsigned)((st + 255) / 256), 256, 0, stream>>>(cenc, st);
  zero_f16<<<(unsigned)((st + 255) / 256), 256, 0, stream>>>(henc16[0], st);
  zero_f32<<<(unsigned)(((long)out_size + 255) / 256), 256, 0, stream>>>(out, (long)out_size);
  set_pred0<<<1, 256, 0, stream>>>(out);

  // Fused layer launcher: grid (8 m-tiles * 16 strips)/4 waves = 32 blocks, z=2
  auto layer = [&](const _Float16* X, int Kx, const _Float16* WxP,
                   const _Float16* h16in, const _Float16* WhP, const float* bias,
                   float* c32s, float* h32s, _Float16* h16out, _Float16* c16out) {
    wmma_lstm_layer<<<dim3(32, 1, 2), 128, 0, stream>>>(
        X, Kx, WxP, (long)G_ * Kx, h16in, BH, WhP, (long)G_ * H_,
        bias, (long)G_, c32s, h32s, h16out, c16out, ybuf, BH);
  };

  // ================= encoder =================
  for (int t = 0; t < TIN_; ++t) {
    const int pi = t & 1, po = (t + 1) & 1;   // h16 double-buffer parity
    embed_tokens<<<256, 256, 0, stream>>>(input + t, TIN_, enc_emb, x16);
    for (int l = 0; l < 3; ++l) {
      const _Float16* X  = l ? ybuf : x16;
      const int       Kx = l ? 512 : E_;
      const _Float16* Wx = l ? eWih1_16 + (size_t)(l - 1) * 2 * G_ * 512 : eWih0_16;
      layer(X, Kx, Wx,
            henc16[pi] + (size_t)(2 * l) * BH, eWhh_16 + (size_t)(2 * l) * G_ * H_,
            enc_b + (size_t)l * 2 * G_,
            cenc + (size_t)(2 * l) * BH, nullptr,
            henc16[po] + (size_t)(2 * l) * BH, cenc16 + (size_t)(2 * l) * BH);
    }
    // attention projection of the *new* hidden state -> encoder_outputs[t]
    gemm(henc16[po], laW1_16, H_, hn, H_, nullptr, la_b1, D_ * B_, H_, H_);
    mix_dirs<<<256, 256, 0, stream>>>(hn, la_W2, la_b2, encout + (size_t)t * D_ * BH, nullptr);
  }
  _Float16* hencF = henc16[TIN_ & 1];   // final encoder hidden (f16)

  // ================= bridge =================
  gemm(hencF, lhW1_16, H_, hn, H_, nullptr, lh_b1, D_ * B_, H_, H_);
  mix_dirs<<<256, 256, 0, stream>>>(hn, lh_W2, lh_b2, hdec, hdec16[0]);
  gemm(cenc16, lcW1_16, H_, hn, H_, nullptr, lc_b1, D_ * B_, H_, H_);
  mix_dirs<<<256, 256, 0, stream>>>(hn, lc_W2, lc_b2, cdec, nullptr);

  // ================= decoder (teacher forced) =================
  for (int t = 0; t < TOUT_ - 1; ++t) {
    const int pi = t & 1, po = (t + 1) & 1;
    attn_ctx<<<256, 256, 0, stream>>>(hdec, encout, ctx);
    build_dec_input<<<512, 256, 0, stream>>>(target, t, dec_emb, ctx, xcat16);
    gemm(xcat16, attnW16, 512, nullptr, E_, xatt16, attn_b, B_, E_, E_);
    for (int l = 0; l < 3; ++l) {
      const _Float16* X  = l ? ybuf : xatt16;
      const int       Kx = l ? 512 : E_;
      const _Float16* Wx = l ? dWih1_16 + (size_t)(l - 1) * 2 * G_ * 512 : dWih0_16;
      layer(X, Kx, Wx,
            hdec16[pi] + (size_t)(2 * l) * BH, dWhh_16 + (size_t)(2 * l) * G_ * H_,
            dec_b + (size_t)l * 2 * G_,
            cdec + (size_t)(2 * l) * BH, hdec + (size_t)(2 * l) * BH,
            hdec16[po] + (size_t)(2 * l) * BH, nullptr);
    }
    // logits -> predicted[t+1]
    gemm(ybuf, fcW16, 512, out + (size_t)(t + 1) * B_ * VDEC_, VDEC_, nullptr, fc_b,
         B_, VDEC_, 128);
  }
}
